// MultiHeadAttention_8203387536069
// MI455X (gfx1250) — matrix operand order repaired and checked by reading, not compiled, not run on hardware
//
#include <hip/hip_runtime.h>
#include <hip/hip_bf16.h>

#define D_MODEL 1024
#define SEQ     2048
#define BATCH   2
#define NHEAD   16
#define DHEAD   64
#define NTOK    (BATCH*SEQ)

typedef __bf16 bf16_t;
typedef __attribute__((ext_vector_type(16))) __bf16        v16bf;
typedef __attribute__((ext_vector_type(8)))  float         v8f;
typedef __attribute__((ext_vector_type(4)))  float         f32x4;
typedef __attribute__((ext_vector_type(2)))  unsigned int  u32x2;
typedef __attribute__((ext_vector_type(4)))  unsigned int  u32x4;
typedef __attribute__((ext_vector_type(8)))  unsigned int  u32x8;

// Load a 16-element bf16 WMMA fragment as two aligned 16-byte chunks.
__device__ __forceinline__ v16bf ld_frag(const bf16_t* base, int off0, int off1) {
  v16bf r;
  ((u32x4*)&r)[0] = *(const u32x4*)(base + off0);
  ((u32x4*)&r)[1] = *(const u32x4*)(base + off1);
  return r;
}

__device__ __forceinline__ v8f wmma_bf16(v16bf a, v16bf b, v8f c) {
  return __builtin_amdgcn_wmma_f32_16x16x32_bf16(false, a, false, b, (short)0, c,
                                                 false, false);
}

// Generic (flat) pointer to a __shared__ object: low 32 bits = LDS byte offset.
#define LDSOFF(p) ((unsigned)(unsigned long long)(p))

// ---------------------------------------------------------------------------
// TDM: DMA a 2D bf16 tile (tile_w x tile_h, tile_w == 64 elems) from global
// into LDS with per-row padding of 16B (row stride 72 bf16 = 144B).
// D# per CDNA5 ISA:
//   group0: count=1 | lds_addr | global_addr[56:0] | type=2
//   group1: data_size=2B, pad_enable, pad_interval=32dw(code 4),
//           pad_amount=4dw(code 3); tensor dims/strides; tile dims.
// All inputs are wave-uniform; issued by one wave, tracked by TENSORcnt.
// ---------------------------------------------------------------------------
__device__ __forceinline__ void tdm_load_2d(const void* gaddr, unsigned lds_off,
                                            unsigned tile_h, unsigned dim0,
                                            unsigned dim1, unsigned stride0)
{
  unsigned long long ga = (unsigned long long)gaddr;
  u32x4 g0;
  g0[0] = 1u;                                   // count=1 (valid), user mode
  g0[1] = lds_off;                              // LDS byte address
  g0[2] = (unsigned)ga;                         // global_addr[31:0]
  g0[3] = ((unsigned)(ga >> 32) & 0x01FFFFFFu)  // global_addr[56:32]
          | 0x80000000u;                        // type=2 ("image")
  u32x8 g1;
  g1[0] = (1u << 16)    // data_size = 1 -> 2 bytes
        | (1u << 20)    // pad_enable
        | (4u << 22)    // pad_interval: 32 dwords (= one 64-elem bf16 row)
        | (3u << 25);   // pad_amount: 4 dwords (16B) -> row stride 144B
  g1[1] = (dim0 & 0xFFFFu) << 16;                       // tensor_dim0[15:0]
  g1[2] = (dim0 >> 16) | ((dim1 & 0xFFFFu) << 16);      // dim0[31:16]|dim1[15:0]
  g1[3] = (dim1 >> 16) | (64u << 16);                   // dim1[31:16]|tile_dim0=64
  g1[4] = tile_h & 0xFFFFu;                             // tile_dim1 (tile_dim2=0)
  g1[5] = stride0;                                      // tensor_dim0_stride[31:0]
  g1[6] = 0u;                                           // stride0 hi | stride1 lo
  g1[7] = 0u;                                           // stride1 hi
  asm volatile("tensor_load_to_lds %0, %1" :: "s"(g0), "s"(g1) : "memory");
}

// ---------------------------------------------------------------------------
// Kernel 1: y = X @ W^T + b (bf16 out, scale fused).
// v_transposed=0: store [B][H][S][Dh]; v_transposed=1: store [B][H][Dh][S].
// Block: 256 threads = 8 waves, tile 128(M) x 64(N), K-step 64, f32x4 staging.
// ---------------------------------------------------------------------------
#define PM 128
#define PN 64
#define PK 64
#define LK 72   // padded bf16 row stride (multiple of 8 -> 16B-aligned chunks)

__global__ __launch_bounds__(256) void qkv_proj_kernel(
    const float* __restrict__ X, const float* __restrict__ W,
    const float* __restrict__ bias, bf16_t* __restrict__ dst,
    float scale, int v_transposed)
{
  __shared__ bf16_t As[PM][LK];
  __shared__ bf16_t Bs[PN][LK];

  const int t    = threadIdx.x;
  const int lane = t & 31;
  const int w    = t >> 5;
  const int wm   = w & 3;      // 4 waves along M (32 rows each)
  const int wn   = w >> 2;     // 2 waves along N (32 cols each)
  const int m0   = blockIdx.y * PM;
  const int n0   = blockIdx.x * PN;

  const int ncol    = lane & 15;
  const int ab0     = (lane < 16) ? 0 : 8;   // A frag K base
  const int bb0     = (lane < 16) ? 0 : 8;   // B frag K base
  const int halfoff = (lane < 16) ? 0 : 8;   // C/D frag row offset

  v8f acc[2][2];
  #pragma unroll
  for (int i = 0; i < 2; ++i)
    #pragma unroll
    for (int j = 0; j < 2; ++j)
      #pragma unroll
      for (int e = 0; e < 8; ++e) acc[i][j][e] = 0.0f;

  for (int kt = 0; kt < D_MODEL; kt += PK) {
    // Stage A: 128x64 f32 via float4 loads, packed bf16 x4 stores (b64).
    #pragma unroll
    for (int i = 0; i < (PM*PK/4)/256; ++i) {   // 8
      int idx = i*256 + t;
      int r = idx >> 4, c4 = idx & 15;          // 16 float4 per row
      f32x4 x = *(const f32x4*)(X + (size_t)(m0 + r)*D_MODEL + kt + c4*4);
      union { bf16_t e[4]; u32x2 u; } cv;
      cv.e[0] = (bf16_t)x[0]; cv.e[1] = (bf16_t)x[1];
      cv.e[2] = (bf16_t)x[2]; cv.e[3] = (bf16_t)x[3];
      *(u32x2*)&As[r][c4*4] = cv.u;
    }
    // Stage B = W rows (y[m,n] = sum_k x[m,k]*W[n,k]).
    #pragma unroll
    for (int i = 0; i < (PN*PK/4)/256; ++i) {   // 4
      int idx = i*256 + t;
      int r = idx >> 4, c4 = idx & 15;
      f32x4 x = *(const f32x4*)(W + (size_t)(n0 + r)*D_MODEL + kt + c4*4);
      union { bf16_t e[4]; u32x2 u; } cv;
      cv.e[0] = (bf16_t)x[0]; cv.e[1] = (bf16_t)x[1];
      cv.e[2] = (bf16_t)x[2]; cv.e[3] = (bf16_t)x[3];
      *(u32x2*)&Bs[r][c4*4] = cv.u;
    }
    __syncthreads();

    #pragma unroll
    for (int ks = 0; ks < 2; ++ks) {
      const int ko = ks*32;
      v16bf a0 = ld_frag(&As[wm*32 +  0 + ncol][0], ko + ab0, ko + ab0 + 16);
      v16bf a1 = ld_frag(&As[wm*32 + 16 + ncol][0], ko + ab0, ko + ab0 + 16);
      v16bf b0 = ld_frag(&Bs[wn*32 +  0 + ncol][0], ko + bb0, ko + bb0 + 16);
      v16bf b1 = ld_frag(&Bs[wn*32 + 16 + ncol][0], ko + bb0, ko + bb0 + 16);
      acc[0][0] = wmma_bf16(a0, b0, acc[0][0]);
      acc[0][1] = wmma_bf16(a0, b1, acc[0][1]);
      acc[1][0] = wmma_bf16(a1, b0, acc[1][0]);
      acc[1][1] = wmma_bf16(a1, b1, acc[1][1]);
    }
    __syncthreads();
  }

  // Epilogue: bias + scale, bf16 store.
  #pragma unroll
  for (int mi = 0; mi < 2; ++mi)
    #pragma unroll
    for (int ni = 0; ni < 2; ++ni) {
      int n  = n0 + wn*32 + ni*16 + ncol;
      float bv = bias[n];
      int h = n >> 6, dh = n & 63;
      if (v_transposed) {
        // [B][H][Dh][S]: rows r are consecutive tokens -> one 16B store.
        int row0 = m0 + wm*32 + mi*16 + halfoff;
        int bb = row0 >> 11, s0 = row0 & (SEQ - 1);
        union { bf16_t e[8]; u32x4 q; } pk;
        #pragma unroll
        for (int r = 0; r < 8; ++r)
          pk.e[r] = (bf16_t)((acc[mi][ni][r] + bv) * scale);
        *(u32x4*)&dst[(((size_t)bb*NHEAD + h)*DHEAD + dh)*SEQ + s0] = pk.q;
      } else {
        #pragma unroll
        for (int r = 0; r < 8; ++r) {
          int row = m0 + wm*32 + mi*16 + r + halfoff;
          int bb = row >> 11, s = row & (SEQ - 1);
          float val = (acc[mi][ni][r] + bv) * scale;
          dst[(((size_t)bb*NHEAD + h)*SEQ + s)*DHEAD + dh] = (bf16_t)val;
        }
      }
    }
}

// ---------------------------------------------------------------------------
// Kernel 2: flash attention per (b, h, 128-row q tile). Key blocks of 64,
// K/V tiles DMA'd by the Tensor Data Mover, double-buffered.
// ---------------------------------------------------------------------------
#define QT    128
#define KTILE 64
#define LS    72

__global__ __launch_bounds__(256) void attn_kernel(
    const bf16_t* __restrict__ Q,  const bf16_t* __restrict__ Kp,
    const bf16_t* __restrict__ Vt_g, const unsigned char* __restrict__ mask,
    float* __restrict__ out)
{
  __shared__ bf16_t Qs[QT][LS];            // [qrow][d]
  __shared__ bf16_t Ks[2][KTILE][LS];      // [key][d]  (B operand of QK^T)
  __shared__ bf16_t Vt[2][DHEAD][LS];      // [d][key]  (B operand of P@V)
  __shared__ bf16_t Ps[8][16][LS];         // per-wave P strip [row][key]

  const int t    = threadIdx.x;
  const int lane = t & 31;
  const int w    = t >> 5;                 // wave owns q rows [w*16, w*16+16)
  const int q0   = blockIdx.x * QT;
  const int h    = blockIdx.y;
  const int b    = blockIdx.z;
  const size_t headbase = ((size_t)b*NHEAD + h) * (size_t)SEQ * DHEAD;

  const int ncol    = lane & 15;
  const int ab0     = (lane < 16) ? 0 : 8;
  const int bb0     = (lane < 16) ? 0 : 8;
  const int halfoff = (lane < 16) ? 0 : 8;

  const bf16_t* qbase = Q    + headbase + (size_t)q0*DHEAD;   // [S][Dh] slice
  const bf16_t* kbase = Kp   + headbase;                      // [S][Dh]
  const bf16_t* vbase = Vt_g + headbase;                      // [Dh][S] (same elems/head)

  // Prologue DMA: Q tile + key-block 0 of K and V. Issued by wave 0 only
  // (TDM ignores EXEC; one descriptor per tile).
  if (w == 0) {
    tdm_load_2d(qbase, LDSOFF(&Qs[0][0]),    QT,    DHEAD, SEQ, DHEAD);
    tdm_load_2d(kbase, LDSOFF(&Ks[0][0][0]), KTILE, DHEAD, SEQ, DHEAD);
    tdm_load_2d(vbase, LDSOFF(&Vt[0][0][0]), DHEAD, SEQ,   DHEAD, SEQ);
  }

  float mrow[8], lrow[8];
  v8f o[4];
  #pragma unroll
  for (int r = 0; r < 8; ++r) { mrow[r] = -3.0e38f; lrow[r] = 0.0f; }
  #pragma unroll
  for (int j = 0; j < 4; ++j)
    #pragma unroll
    for (int r = 0; r < 8; ++r) o[j][r] = 0.0f;

  const int nblk = SEQ / KTILE;            // 32
  for (int ib = 0; ib < nblk; ++ib) {
    const int buf = ib & 1;
    const int kb  = ib * KTILE;
    if (w == 0) {
      if (ib + 1 < nblk) {
        // Issue next block's DMA, then retire the two oldest transfers
        // (this block's K and V; TDM completes in-order per wave).
        const size_t kb2 = (size_t)(ib + 1) * KTILE;
        tdm_load_2d(kbase + kb2*DHEAD, LDSOFF(&Ks[buf^1][0][0]),
                    KTILE, DHEAD, SEQ, DHEAD);
        tdm_load_2d(vbase + kb2,       LDSOFF(&Vt[buf^1][0][0]),
                    DHEAD, SEQ, DHEAD, SEQ);
        __builtin_amdgcn_s_wait_tensorcnt(2);
      } else {
        __builtin_amdgcn_s_wait_tensorcnt(0);
      }
    }
    __syncthreads();

    // Scores: S = Q(16x64) @ K^T, two K-depth-32 WMMA steps per 16-col tile.
    v8f s[4];
    #pragma unroll
    for (int j = 0; j < 4; ++j)
      #pragma unroll
      for (int r = 0; r < 8; ++r) s[j][r] = 0.0f;
    #pragma unroll
    for (int kstep = 0; kstep < 2; ++kstep) {
      v16bf aq = ld_frag(&Qs[w*16 + ncol][0], kstep*32 + ab0, kstep*32 + ab0 + 16);
      #pragma unroll
      for (int j = 0; j < 4; ++j) {
        v16bf bk = ld_frag(&Ks[buf][j*16 + ncol][0],
                           kstep*32 + bb0, kstep*32 + bb0 + 16);
        s[j] = wmma_bf16(aq, bk, s[j]);
      }
    }

    if (kb + KTILE < SEQ) {  // warm the cache for next block's mask bytes
      __builtin_prefetch(mask + ((size_t)b*SEQ + q0 + t)*SEQ + kb + KTILE, 0, 1);
    }

    // Mask + online softmax (row stats reduced over the 16-lane N dimension).
    #pragma unroll
    for (int r = 0; r < 8; ++r) {
      const int grow = q0 + w*16 + r + halfoff;
      const unsigned char* mp = mask + ((size_t)b*SEQ + grow)*SEQ + kb;
      float sv[4];
      float vmax = -3.0e38f;
      #pragma unroll
      for (int j = 0; j < 4; ++j) {
        float x = s[j][r];
        if (mp[j*16 + ncol]) x = -1.0e30f;   // mask==true -> MASK_VAL
        sv[j] = x;
        vmax = fmaxf(vmax, x);
      }
      #pragma unroll
      for (int d = 1; d < 16; d <<= 1) vmax = fmaxf(vmax, __shfl_xor(vmax, d, 32));
      const float mnew  = fmaxf(mrow[r], vmax);
      const float alpha = __expf(mrow[r] - mnew);
      float pj[4];
      float rs = 0.0f;
      #pragma unroll
      for (int j = 0; j < 4; ++j) { pj[j] = __expf(sv[j] - mnew); rs += pj[j]; }
      #pragma unroll
      for (int d = 1; d < 16; d <<= 1) rs += __shfl_xor(rs, d, 32);
      lrow[r] = lrow[r]*alpha + rs;
      mrow[r] = mnew;
      #pragma unroll
      for (int j = 0; j < 4; ++j) o[j][r] *= alpha;
      // Re-layout P (C-layout f32) -> A-layout bf16 via per-wave LDS strip.
      #pragma unroll
      for (int j = 0; j < 4; ++j) Ps[w][r + halfoff][j*16 + ncol] = (bf16_t)pj[j];
    }
    // Per-wave LDS traffic is in-order; reads below see the stores above.

    // O += P(16 x 64 keys) @ V(64 keys x 64 d)
    #pragma unroll
    for (int kstep = 0; kstep < 2; ++kstep) {
      v16bf ap = ld_frag(&Ps[w][ncol][0], kstep*32 + ab0, kstep*32 + ab0 + 16);
      #pragma unroll
      for (int j = 0; j < 4; ++j) {
        v16bf bv = ld_frag(&Vt[buf][j*16 + ncol][0],
                           kstep*32 + bb0, kstep*32 + bb0 + 16);
        o[j] = wmma_bf16(ap, bv, o[j]);
      }
    }
    __syncthreads();
  }

  // Normalize and store merged-head f32 output [B][S][D_MODEL].
  #pragma unroll
  for (int j = 0; j < 4; ++j) {
    const int d = j*16 + ncol;
    #pragma unroll
    for (int r = 0; r < 8; ++r) {
      const int grow = q0 + w*16 + r + halfoff;
      out[((size_t)b*SEQ + grow)*D_MODEL + h*DHEAD + d] = o[j][r] / lrow[r];
    }
  }
}

// ---------------------------------------------------------------------------
extern "C" void kernel_launch(void* const* d_in, const int* in_sizes, int n_in,
                              void* d_out, int out_size, void* d_ws, size_t ws_size,
                              hipStream_t stream) {
  const float* query = (const float*)d_in[0];
  const float* key   = (const float*)d_in[1];
  const float* value = (const float*)d_in[2];
  const unsigned char* mask = (const unsigned char*)d_in[3];  // jax bool = 1B
  const float* Wq = (const float*)d_in[4];
  const float* bq = (const float*)d_in[5];
  const float* Wk = (const float*)d_in[6];
  const float* bk = (const float*)d_in[7];
  const float* Wv = (const float*)d_in[8];
  const float* bv = (const float*)d_in[9];

  const size_t per = (size_t)BATCH * NHEAD * SEQ * DHEAD;  // 4M elems, 8MB bf16
  bf16_t* qws = (bf16_t*)d_ws;
  bf16_t* kws = qws + per;
  bf16_t* vws = kws + per;   // stored [B][H][Dh][S]

  dim3 blk(256);
  dim3 pg(D_MODEL / PN, NTOK / PM);   // (16, 32)
  qkv_proj_kernel<<<pg, blk, 0, stream>>>(query, Wq, bq, qws, 0.125f, 0);
  qkv_proj_kernel<<<pg, blk, 0, stream>>>(key,   Wk, bk, kws, 1.0f,   0);
  qkv_proj_kernel<<<pg, blk, 0, stream>>>(value, Wv, bv, vws, 1.0f,   1);

  dim3 ag(SEQ / QT, NHEAD, BATCH);    // (16, 16, 2)
  attn_kernel<<<ag, blk, 0, stream>>>(qws, kws, vws, mask, (float*)d_out);

  (void)in_sizes; (void)n_in; (void)out_size; (void)ws_size;
}
